// ModuleNet_36232344109729
// MI455X (gfx1250) — matrix operand order repaired and checked by reading, compiled, not run on hardware
//
#include <hip/hip_runtime.h>
#include <hip/hip_bf16.h>
#include <math.h>

#define E_DIM 128
#define M_MOD 16
#define B_PATH 2048
#define L_STEP 8
#define H_DIM 256
#define O_DIM 8

typedef __bf16 bf16s;
typedef bf16s v16bf __attribute__((ext_vector_type(16)));
typedef float v8f  __attribute__((ext_vector_type(8)));
typedef unsigned int v8u __attribute__((ext_vector_type(8)));

__device__ __forceinline__ unsigned short f2bf(float f) {
    unsigned int u = __float_as_uint(f);
    u += 0x7FFFu + ((u >> 16) & 1u);   // round-to-nearest-even
    return (unsigned short)(u >> 16);
}

__device__ __forceinline__ v16bf make_frag(uint4 lo, uint4 hi) {
    v8u t;
    t[0] = lo.x; t[1] = lo.y; t[2] = lo.z; t[3] = lo.w;
    t[4] = hi.x; t[5] = hi.y; t[6] = hi.z; t[7] = hi.w;
    return __builtin_bit_cast(v16bf, t);
}

// ---------------- prep kernels ----------------

__global__ void k_zero_counts(int* counts) {
    counts[threadIdx.x] = 0;            // blockDim.x == L*M == 128
}

__global__ void k_bucket(const int* __restrict__ module_ids,
                         int* __restrict__ counts, int* __restrict__ idx_buf) {
    int idx = blockIdx.x * blockDim.x + threadIdx.x;
    if (idx >= B_PATH * L_STEP) return;
    int p = idx / L_STEP;
    int l = idx % L_STEP;
    int m = module_ids[p * L_STEP + l];
    int slot = atomicAdd(&counts[l * M_MOD + m], 1);
    idx_buf[(l * M_MOD + m) * B_PATH + slot] = p;
}

__global__ void k_gather_x0(const float* __restrict__ author_table,
                            const int* __restrict__ author_ids,
                            float* __restrict__ x) {
    int idx = blockIdx.x * blockDim.x + threadIdx.x;  // B*E threads
    int p = idx >> 7;
    int e = idx & (E_DIM - 1);
    x[idx] = author_table[(long)author_ids[p] * E_DIM + e];
}

// W_mod [M][2E][E] f32 -> WbfT [M][E(n)][2E(k)] bf16 ; Wc1 [E][H] f32 -> Wc1T [H(n)][E(k)] bf16
__global__ void k_convert_weights(const float* __restrict__ W_mod,
                                  const float* __restrict__ Wc1,
                                  unsigned short* __restrict__ WbfT,
                                  unsigned short* __restrict__ Wc1T) {
    int idx = blockIdx.x * blockDim.x + threadIdx.x;
    const int NW = M_MOD * E_DIM * 2 * E_DIM;         // 524288
    if (idx < NW) {
        int m   = idx / (E_DIM * 2 * E_DIM);
        int rem = idx % (E_DIM * 2 * E_DIM);
        int n   = rem / (2 * E_DIM);
        int k   = rem % (2 * E_DIM);
        WbfT[idx] = f2bf(W_mod[((long)m * (2 * E_DIM) + k) * E_DIM + n]);
    } else {
        int j = idx - NW;
        if (j < H_DIM * E_DIM) {
            int n = j / E_DIM;
            int k = j % E_DIM;
            Wc1T[j] = f2bf(Wc1[(long)k * H_DIM + n]);
        }
    }
}

// ---------------- recurrent step: per-(module) dense GEMM tiles via WMMA ----------------

__global__ void __launch_bounds__(256)
k_step(const float* __restrict__ x_cur, float* __restrict__ x_next,
       const float* __restrict__ node_table, const int* __restrict__ node_ids,
       const unsigned short* __restrict__ WbfT, const float* __restrict__ b_mod,
       const int* __restrict__ idx_buf, const int* __restrict__ counts, int l)
{
    __shared__ __align__(16) unsigned short catA[16][2 * E_DIM];  // 16 rows x 256 bf16
    __shared__ int paths[16];

    int m = blockIdx.y;
    int cnt = counts[l * M_MOD + m];
    int base = blockIdx.x * 16;
    if (base >= cnt) return;                 // uniform early-exit

    int tid = threadIdx.x;
    if (tid < 16) {
        int slot = base + tid;
        paths[tid] = (slot < cnt) ? idx_buf[(l * M_MOD + m) * B_PATH + slot] : -1;
    }
    __syncthreads();

    // stage cat = [x | node_emb] rows as bf16 into LDS (16 threads per row, 16 elems each)
    {
        int r  = tid >> 4;
        int c0 = (tid & 15) * 16;
        int p  = paths[r];
        if (p >= 0) {
            if (c0 < E_DIM) {
                const float* src = x_cur + (long)p * E_DIM + c0;
                #pragma unroll
                for (int k = 0; k < 16; ++k) catA[r][c0 + k] = f2bf(src[k]);
            } else {
                int nid = node_ids[p * L_STEP + l];
                const float* src = node_table + (long)nid * E_DIM + (c0 - E_DIM);
                #pragma unroll
                for (int k = 0; k < 16; ++k) catA[r][c0 + k] = f2bf(src[k]);
            }
        } else {
            #pragma unroll
            for (int k = 0; k < 16; ++k) catA[r][c0 + k] = 0;
        }
    }
    __syncthreads();

    int wave  = tid >> 5;           // 8 waves: one 16-col output tile each
    int lane  = tid & 31;
    int nloc  = wave * 16 + (lane & 15);
    int mrow  = lane & 15;
    int khalf = lane >> 4;

    const unsigned short* wrow = WbfT + ((long)m * E_DIM + nloc) * (2 * E_DIM);

    v8f acc = {};
    #pragma unroll
    for (int kc = 0; kc < 8; ++kc) {
        uint4 alo = *(const uint4*)&catA[mrow][kc * 32 + khalf * 16];
        uint4 ahi = *(const uint4*)&catA[mrow][kc * 32 + khalf * 16 + 8];
        v16bf a = make_frag(alo, ahi);
        const uint4* pb = (const uint4*)(wrow + kc * 32 + khalf * 16);
        v16bf b = make_frag(pb[0], pb[1]);
        acc = __builtin_amdgcn_wmma_f32_16x16x32_bf16(false, a, false, b,
                                                      (short)0, acc, false, false);
    }

    float bias = b_mod[m * E_DIM + nloc];
    #pragma unroll
    for (int rr = 0; rr < 8; ++rr) {
        int slotM = khalf * 8 + rr;          // C layout: lanes 16-31 hold rows 8..15
        int pp = paths[slotM];
        if (pp >= 0) x_next[(long)pp * E_DIM + nloc] = tanhf(acc[rr] + bias);
    }
}

// ---------------- classifier stage 1: h = tanh(x @ Wc1 + bc1), WMMA ----------------

__global__ void __launch_bounds__(256)
k_cls1(const float* __restrict__ x, const unsigned short* __restrict__ Wc1T,
       const float* __restrict__ bc1, float* __restrict__ h)
{
    __shared__ __align__(16) unsigned short catX[16][E_DIM];
    int tid  = threadIdx.x;
    int row0 = blockIdx.x * 16;
    {
        int r  = tid >> 4;
        int c0 = (tid & 15) * 8;
        const float* src = x + (long)(row0 + r) * E_DIM + c0;
        #pragma unroll
        for (int k = 0; k < 8; ++k) catX[r][c0 + k] = f2bf(src[k]);
    }
    __syncthreads();

    int wave  = tid >> 5;
    int lane  = tid & 31;
    int nglob = blockIdx.y * 128 + wave * 16 + (lane & 15);
    int mrow  = lane & 15;
    int khalf = lane >> 4;
    const unsigned short* wrow = Wc1T + (long)nglob * E_DIM;

    v8f acc = {};
    #pragma unroll
    for (int kc = 0; kc < 4; ++kc) {
        uint4 alo = *(const uint4*)&catX[mrow][kc * 32 + khalf * 16];
        uint4 ahi = *(const uint4*)&catX[mrow][kc * 32 + khalf * 16 + 8];
        v16bf a = make_frag(alo, ahi);
        const uint4* pb = (const uint4*)(wrow + kc * 32 + khalf * 16);
        v16bf b = make_frag(pb[0], pb[1]);
        acc = __builtin_amdgcn_wmma_f32_16x16x32_bf16(false, a, false, b,
                                                      (short)0, acc, false, false);
    }

    float bias = bc1[nglob];
    #pragma unroll
    for (int rr = 0; rr < 8; ++rr) {
        int grow = row0 + khalf * 8 + rr;
        h[(long)grow * H_DIM + nglob] = tanhf(acc[rr] + bias);
    }
}

// ---------------- classifier stage 2: out = h @ Wc2 + bc2 (tiny, VALU) ----------------

__global__ void k_cls2(const float* __restrict__ h, const float* __restrict__ Wc2,
                       const float* __restrict__ bc2, float* __restrict__ out)
{
    int idx = blockIdx.x * blockDim.x + threadIdx.x;   // B*O
    int p = idx >> 3;
    int o = idx & (O_DIM - 1);
    const float* hr = h + (long)p * H_DIM;
    float s = bc2[o];
    #pragma unroll 4
    for (int k = 0; k < H_DIM; ++k) s += hr[k] * Wc2[k * O_DIM + o];
    out[idx] = s;
}

// ---------------- host launcher ----------------

extern "C" void kernel_launch(void* const* d_in, const int* in_sizes, int n_in,
                              void* d_out, int out_size, void* d_ws, size_t ws_size,
                              hipStream_t stream) {
    const float* author_table = (const float*)d_in[0];
    const float* node_table   = (const float*)d_in[1];
    const float* W_mod        = (const float*)d_in[2];
    const float* b_mod        = (const float*)d_in[3];
    const float* Wc1          = (const float*)d_in[4];
    const float* bc1          = (const float*)d_in[5];
    const float* Wc2          = (const float*)d_in[6];
    const float* bc2          = (const float*)d_in[7];
    const int* author_ids     = (const int*)d_in[8];
    const int* module_ids     = (const int*)d_in[9];
    const int* node_ids       = (const int*)d_in[10];
    float* out = (float*)d_out;

    char* ws = (char*)d_ws;
    float* xa            = (float*)(ws);                                    // 1 MB
    float* xb            = (float*)(ws + (1u << 20));                       // 1 MB
    unsigned short* WbfT = (unsigned short*)(ws + 2u * (1u << 20));         // 1 MB
    unsigned short* Wc1T = (unsigned short*)(ws + 3u * (1u << 20));         // 64 KB
    int* idx_buf         = (int*)(ws + 3u * (1u << 20) + (1u << 16));       // 1 MB
    int* counts          = (int*)(ws + 4u * (1u << 20) + (1u << 16));       // 512 B
    float* hbuf          = (float*)(ws + 4u * (1u << 20) + (1u << 16) + 4096); // 2 MB

    hipLaunchKernelGGL(k_zero_counts, dim3(1), dim3(L_STEP * M_MOD), 0, stream, counts);
    hipLaunchKernelGGL(k_bucket, dim3((B_PATH * L_STEP + 255) / 256), dim3(256), 0, stream,
                       module_ids, counts, idx_buf);
    hipLaunchKernelGGL(k_gather_x0, dim3(B_PATH * E_DIM / 256), dim3(256), 0, stream,
                       author_table, author_ids, xa);
    int totW = M_MOD * E_DIM * 2 * E_DIM + H_DIM * E_DIM;
    hipLaunchKernelGGL(k_convert_weights, dim3((totW + 255) / 256), dim3(256), 0, stream,
                       W_mod, Wc1, WbfT, Wc1T);

    float* xc = xa;
    float* xn = xb;
    for (int l = 0; l < L_STEP; ++l) {
        hipLaunchKernelGGL(k_step, dim3(B_PATH / 16, M_MOD), dim3(256), 0, stream,
                           xc, xn, node_table, node_ids, WbfT, b_mod, idx_buf, counts, l);
        float* t = xc; xc = xn; xn = t;
    }

    hipLaunchKernelGGL(k_cls1, dim3(B_PATH / 16, H_DIM / 128), dim3(256), 0, stream,
                       xc, Wc1T, bc1, hbuf);
    hipLaunchKernelGGL(k_cls2, dim3(B_PATH * O_DIM / 256), dim3(256), 0, stream,
                       hbuf, Wc2, bc2, out);
}
